// EncoderLayer_33054068310189
// MI455X (gfx1250) — matrix operand order repaired and checked by reading, not compiled, not run on hardware
//
#include <hip/hip_runtime.h>
#include <hip/hip_bf16.h>

// ---------------------------------------------------------------------------
// CDNA5 (gfx1250) bf16-WMMA transformer encoder layer.
// Matrix ops: v_wmma_f32_16x16x32_bf16.  LDS staging of A-side tiles via the
// Tensor Data Mover (tensor_load_to_lds + s_wait_tensorcnt), with D#-encoded
// LDS row padding reproducing the bank-conflict-free 112B stride.
// ---------------------------------------------------------------------------

typedef __attribute__((ext_vector_type(16))) __bf16 bf16x16;
typedef __attribute__((ext_vector_type(8)))  __bf16 bf16x8;
typedef __attribute__((ext_vector_type(8)))  float  f32x8;
typedef __attribute__((ext_vector_type(4)))  unsigned int u32x4;
typedef __attribute__((ext_vector_type(8)))  int i32x8;
typedef __attribute__((ext_vector_type(4)))  int i32x4;

constexpr int Bc = 4, Lc = 1024, Dc = 1024, Hc = 16, DHc = 64, FFc = 4096;
constexpr int Mrows = Bc * Lc;          // 4096 token rows
constexpr int LSTR  = 56;               // LDS row stride (elems): 112B, 16B-aligned, conflict-free

#if __has_builtin(__builtin_amdgcn_tensor_load_to_lds) && \
    __has_builtin(__builtin_amdgcn_s_wait_tensorcnt)
#define HAVE_TDM 1
#else
#define HAVE_TDM 0
#endif

#if HAVE_TDM
// 2D tile DMA: global (row-major, d0 = contiguous dim, extents in elements of
// 2 bytes) -> LDS at lds_addr, with optional LDS row padding.
// pad_interval_enc: 0=2,1=4,2=8,3=16,4=32... DWORDs between pads.
// pad_amount_dw:    padding in DWORDs (1..128), 0 = no padding.
__device__ __forceinline__ void tdm_load_2d_bf16(
    unsigned lds_addr, const void* gptr,
    unsigned tensor_d0, unsigned tensor_d1, unsigned long long d0_stride,
    unsigned tile_d0, unsigned tile_d1,
    unsigned pad_interval_enc, unsigned pad_amount_dw) {
  unsigned long long ga = (unsigned long long)(uintptr_t)gptr;
  // D# group 0: count=1 (valid user descriptor), lds_addr, global_addr, type=2
  u32x4 g0 = {
    1u,
    lds_addr,
    (unsigned)(ga & 0xffffffffu),
    (unsigned)((ga >> 32) & 0x1ffffffu) | (2u << 30)
  };
  // D# group 1
  unsigned w0 = (1u << 16);  // data_size = 1 -> 2 bytes (bf16)
  if (pad_amount_dw)
    w0 |= (1u << 20) | (pad_interval_enc << 22) | ((pad_amount_dw - 1u) << 25);
  i32x8 g1 = {
    (int)w0,
    (int)((tensor_d0 & 0xffffu) << 16),                                   // dim0[15:0]
    (int)(((tensor_d0 >> 16) & 0xffffu) | ((tensor_d1 & 0xffffu) << 16)), // dim0[31:16], dim1[15:0]
    (int)(((tensor_d1 >> 16) & 0xffffu) | ((tile_d0 & 0xffffu) << 16)),   // dim1[31:16], tile_dim0
    (int)(tile_d1 & 0xffffu),                                             // tile_dim1, tile_dim2=0
    (int)(unsigned)(d0_stride & 0xffffffffull),                           // dim0_stride[31:0]
    (int)(unsigned)((d0_stride >> 32) & 0xffffull),                       // dim0_stride[47:32]
    0
  };
  i32x4 z4 = {0, 0, 0, 0};
#if __clang_major__ >= 23
  i32x8 z8 = {0, 0, 0, 0, 0, 0, 0, 0};
  __builtin_amdgcn_tensor_load_to_lds(g0, g1, z4, z4, z8, 0);
#else
  __builtin_amdgcn_tensor_load_to_lds(g0, g1, z4, z4, 0);
#endif
}
#endif  // HAVE_TDM

__device__ __forceinline__ unsigned lds_offset_of(const void* p) {
  // flat aperture: LDS address = low 32 bits of the generic pointer
  return (unsigned)(uintptr_t)p;
}

__device__ __forceinline__ bf16x16 frag_cat(bf16x8 lo, bf16x8 hi) {
  return __builtin_shufflevector(lo, hi, 0,1,2,3,4,5,6,7,8,9,10,11,12,13,14,15);
}

// A-fragment: 16x32 (MxK). lane m=λ&15; k chunks [kofs,kofs+8) / [16+kofs,...)
__device__ __forceinline__ bf16x16 load_a_frag(const __bf16* base, int stride, int lane) {
  int m    = lane & 15;
  int kofs = (lane >> 4) * 8;
  const __bf16* p = base + m * stride + kofs;
  bf16x8 lo = *(const bf16x8*)(p);
  bf16x8 hi = *(const bf16x8*)(p + 16);
  return frag_cat(lo, hi);
}

// B-fragment: 32x16 (KxN), stored TRANSPOSED in LDS as [n][k].
__device__ __forceinline__ bf16x16 load_b_frag(const __bf16* baseNK, int stride, int lane) {
  int n  = lane & 15;
  int kb = (lane >> 4) * 8;
  const __bf16* p = baseNK + n * stride + kb;
  bf16x8 lo = *(const bf16x8*)(p);
  bf16x8 hi = *(const bf16x8*)(p + 16);
  return frag_cat(lo, hi);
}

__device__ __forceinline__ f32x8 wmma_bf16(bf16x16 a, bf16x16 b, f32x8 c) {
  return __builtin_amdgcn_wmma_f32_16x16x32_bf16(false, a, false, b, (short)0, c, false, false);
}

// ---------------------------------------------------------------------------
// fp32 -> bf16 conversion
// ---------------------------------------------------------------------------
__global__ __launch_bounds__(256) void f32_to_bf16_kernel(const float* __restrict__ in,
                                                          __bf16* __restrict__ out, int n) {
  int i = blockIdx.x * 256 + threadIdx.x;
  if (i < n) out[i] = (__bf16)in[i];
}

// ---------------------------------------------------------------------------
// Generic bf16 GEMM: C = A[M,K] x W[K,N] + bias, fused epilogues.
// Block: 256 threads = 8 waves as 4(M) x 2(N); block tile 64x128; wave tile 16x64.
// A tile (64x32) staged by the Tensor Data Mover with padded LDS rows.
// ---------------------------------------------------------------------------
constexpr int EPI_QKV = 0;       // out bf16 to head-split [B,H,L,DH]
constexpr int EPI_F32RES = 1;    // out f32 = acc + bias + res (residual add)
constexpr int EPI_RELU_BF16 = 2; // out bf16 = relu(acc + bias)

template <int EPI>
__global__ __launch_bounds__(256) void gemm_bf16_kernel(
    const __bf16* __restrict__ A, const __bf16* __restrict__ W,
    const float* __restrict__ bias, const float* __restrict__ res,
    void* __restrict__ outp, int M, int N, int K) {
  __shared__ __bf16 sA[64 * LSTR];
  __shared__ __bf16 sB[128 * LSTR];

  const int tid = threadIdx.x;
  const int lane = tid & 31, wave = tid >> 5;
  const int mw = wave >> 1, nw = wave & 1;
  const int m0 = blockIdx.y * 64, n0 = blockIdx.x * 128;

  f32x8 acc[4] = {};

  const int nsteps = K >> 5;
  for (int kt = 0; kt < nsteps; ++kt) {
    // ---- stage A tile 64x32 into padded LDS rows (stride 56 elems) ----
#if HAVE_TDM
    if (tid == 0) {
      // tile row = 32 bf16 = 16 DWORDs (pad_interval_enc 3); pad 12 DWORDs
      tdm_load_2d_bf16(lds_offset_of(sA), &A[(size_t)m0 * K + kt * 32],
                       (unsigned)K, (unsigned)M, (unsigned long long)K,
                       32u, 64u, 3u, 12u);
    }
#else
    {
      int row = tid >> 2, c8 = (tid & 3) * 8;
      const __bf16* src = &A[(size_t)(m0 + row) * K + (kt * 32 + c8)];
      *(bf16x8*)&sA[row * LSTR + c8] = *(const bf16x8*)src;
    }
#endif
    // ---- stage B tile 32x128, transposed into [n][k] ----
    {
      int kk = tid >> 3, c0 = (tid & 7) * 16;
      const __bf16* src = &W[(size_t)(kt * 32 + kk) * N + (n0 + c0)];
      bf16x8 v0 = *(const bf16x8*)(src);
      bf16x8 v1 = *(const bf16x8*)(src + 8);
      if (kt + 1 < nsteps) __builtin_prefetch(src + (size_t)32 * N, 0, 1);
#pragma unroll
      for (int i = 0; i < 8; ++i) {
        sB[(c0 + i) * LSTR + kk]     = v0[i];
        sB[(c0 + 8 + i) * LSTR + kk] = v1[i];
      }
    }
#if HAVE_TDM
    if (tid == 0) __builtin_amdgcn_s_wait_tensorcnt((short)0);
#endif
    __syncthreads();

    bf16x16 af = load_a_frag(&sA[(mw * 16) * LSTR], LSTR, lane);
#pragma unroll
    for (int j = 0; j < 4; ++j) {
      bf16x16 bfm = load_b_frag(&sB[(nw * 64 + j * 16) * LSTR], LSTR, lane);
      acc[j] = wmma_bf16(af, bfm, acc[j]);
    }
    __syncthreads();
  }

  // epilogue: D frag element (r, λ) -> row r + 8*(λ>=16), col λ&15
  const int nl = lane & 15, mh = (lane >> 4) * 8;
#pragma unroll
  for (int j = 0; j < 4; ++j) {
#pragma unroll
    for (int r = 0; r < 8; ++r) {
      int gr = m0 + mw * 16 + mh + r;
      int gc = n0 + nw * 64 + j * 16 + nl;
      float v = acc[j][r] + bias[gc];
      if constexpr (EPI == EPI_QKV) {
        int b = gr >> 10, l = gr & 1023, hh = gc >> 6, dh = gc & 63;
        ((__bf16*)outp)[(((size_t)(b * Hc + hh)) * Lc + l) * DHc + dh] = (__bf16)v;
      } else if constexpr (EPI == EPI_F32RES) {
        size_t idx = (size_t)gr * N + gc;
        ((float*)outp)[idx] = v + res[idx];
      } else {
        ((__bf16*)outp)[(size_t)gr * N + gc] = (__bf16)fmaxf(v, 0.0f);
      }
    }
  }
}

// ---------------------------------------------------------------------------
// Flash attention: per block = one (b,h) and 128 query rows (8 waves x 16).
// KV tiles of 32 keys in LDS; S and P*V via WMMA; online softmax with
// half-wave (16-lane) shuffle reductions (wave32). K tile staged by TDM.
// score = (q.k / 8) * attn_bias[b,q,k];  masked keys -> -10000 (pre-softmax).
// ---------------------------------------------------------------------------
__global__ __launch_bounds__(256) void flash_attn_kernel(
    const __bf16* __restrict__ Q, const __bf16* __restrict__ Kb,
    const __bf16* __restrict__ Vb, const float* __restrict__ bias,
    const int* __restrict__ mask, __bf16* __restrict__ ctx_out) {
  __shared__ __bf16 sK[32 * 72];        // [key][dh], stride 72 elems (144B)
  __shared__ __bf16 sV[64 * LSTR];      // transposed [dh][key]
  __shared__ __bf16 sP[8 * 16 * LSTR];  // per-wave P tiles (A-operand staging)

  const int tid = threadIdx.x, lane = tid & 31, wave = tid >> 5;
  const int b = blockIdx.z, h = blockIdx.y;
  const int bh = b * Hc + h;
  const int q0 = blockIdx.x * 128 + wave * 16;
  const int nl = lane & 15, mh = (lane >> 4) * 8;

  // Q fragments: DH=64 -> 2 A-frags, loaded once from global
  bf16x16 qf[2];
  {
    int m = lane & 15;
    int ko = (lane >> 4) * 8;
    const __bf16* qp = Q + ((size_t)bh * Lc + q0 + m) * DHc;
#pragma unroll
    for (int ks = 0; ks < 2; ++ks) {
      bf16x8 lo = *(const bf16x8*)(qp + ks * 32 + ko);
      bf16x8 hi = *(const bf16x8*)(qp + ks * 32 + 16 + ko);
      qf[ks] = frag_cat(lo, hi);
    }
  }

  f32x8 ctx[4] = {};
  float mrow[8], lrow[8];
#pragma unroll
  for (int r = 0; r < 8; ++r) { mrow[r] = -1e30f; lrow[r] = 0.0f; }

  for (int kb0 = 0; kb0 < Lc; kb0 += 32) {
    // ---- stage K tile [32 keys][64 dh] (TDM: row 32 DW, pad 4 DW -> stride 72) ----
#if HAVE_TDM
    if (tid == 0) {
      tdm_load_2d_bf16(lds_offset_of(sK), &Kb[((size_t)bh * Lc + kb0) * DHc],
                       (unsigned)DHc, (unsigned)Lc, (unsigned long long)DHc,
                       64u, 32u, 4u, 4u);
    }
#else
    {
      int key = tid >> 3, c8 = (tid & 7) * 8;
      *(bf16x8*)&sK[key * 72 + c8] =
          *(const bf16x8*)&Kb[((size_t)bh * Lc + kb0 + key) * DHc + c8];
    }
#endif
    // ---- stage V tile transposed -> sV[dh][key] ----
    {
      int key = tid >> 3, c8 = (tid & 7) * 8;
      bf16x8 v = *(const bf16x8*)&Vb[((size_t)bh * Lc + kb0 + key) * DHc + c8];
#pragma unroll
      for (int i = 0; i < 8; ++i) sV[(c8 + i) * LSTR + key] = v[i];
    }
#if HAVE_TDM
    if (tid == 0) __builtin_amdgcn_s_wait_tensorcnt((short)0);
#endif
    __syncthreads();

    // S = Q x K^T  (two 16-key subtiles, two dh k-steps each)
    f32x8 S[2] = {};
#pragma unroll
    for (int j = 0; j < 2; ++j) {
#pragma unroll
      for (int ks = 0; ks < 2; ++ks) {
        bf16x16 bfk = load_b_frag(&sK[(j * 16) * 72 + ks * 32], 72, lane);
        S[j] = wmma_bf16(qf[ks], bfk, S[j]);
      }
    }

    // scale * multiplicative bias, mask, online softmax
    int mk0 = mask[b * Lc + kb0 + nl];
    int mk1 = mask[b * Lc + kb0 + 16 + nl];
#pragma unroll
    for (int r = 0; r < 8; ++r) {
      int qrow = q0 + mh + r;
      const float* bp = bias + ((size_t)b * Lc + qrow) * Lc + kb0;
      float s0 = S[0][r] * 0.125f * bp[nl];
      float s1 = S[1][r] * 0.125f * bp[16 + nl];
      if (mk0 == 0) s0 = -10000.0f;
      if (mk1 == 0) s1 = -10000.0f;

      float mx = fmaxf(s0, s1);
#pragma unroll
      for (int off = 1; off < 16; off <<= 1) mx = fmaxf(mx, __shfl_xor(mx, off, 32));
      float mnew  = fmaxf(mrow[r], mx);
      float alpha = __expf(mrow[r] - mnew);
      mrow[r] = mnew;

      float p0 = __expf(s0 - mnew);
      float p1 = __expf(s1 - mnew);
      float ps = p0 + p1;
#pragma unroll
      for (int off = 1; off < 16; off <<= 1) ps += __shfl_xor(ps, off, 32);
      lrow[r] = lrow[r] * alpha + ps;
#pragma unroll
      for (int c = 0; c < 4; ++c) ctx[c][r] *= alpha;

      sP[(wave * 16 + mh + r) * LSTR + nl]      = (__bf16)p0;
      sP[(wave * 16 + mh + r) * LSTR + 16 + nl] = (__bf16)p1;
    }

    // ctx += P x V   (K=32 keys = one WMMA k-step)
    bf16x16 pf = load_a_frag(&sP[(wave * 16) * LSTR], LSTR, lane);
#pragma unroll
    for (int c = 0; c < 4; ++c) {
      bf16x16 vf = load_b_frag(&sV[(c * 16) * LSTR], LSTR, lane);
      ctx[c] = wmma_bf16(pf, vf, ctx[c]);
    }
    __syncthreads();
  }

  // normalize and write merged-heads bf16 context [B,L,D]
#pragma unroll
  for (int c = 0; c < 4; ++c) {
#pragma unroll
    for (int r = 0; r < 8; ++r) {
      int qrow = q0 + mh + r;
      int d = h * DHc + c * 16 + nl;
      float o = ctx[c][r] / lrow[r];
      ctx_out[((size_t)b * Lc + qrow) * Dc + d] = (__bf16)o;
    }
  }
}

// ---------------------------------------------------------------------------
// LayerNorm over D=1024; one block (256 thr, 4 elems each) per token row.
// ---------------------------------------------------------------------------
template <bool WRITE_BF16>
__global__ __launch_bounds__(256) void layernorm_kernel(
    const float* __restrict__ in, const float* __restrict__ gamma,
    const float* __restrict__ beta, float* __restrict__ outf,
    __bf16* __restrict__ outb) {
  __shared__ float red[256];
  const int row = blockIdx.x, tid = threadIdx.x;
  const float* p = in + (size_t)row * Dc + tid * 4;
  float4 v = *(const float4*)p;

  red[tid] = v.x + v.y + v.z + v.w;
  __syncthreads();
  for (int st = 128; st > 0; st >>= 1) {
    if (tid < st) red[tid] += red[tid + st];
    __syncthreads();
  }
  float mean = red[0] * (1.0f / Dc);
  __syncthreads();

  float dx = v.x - mean, dy = v.y - mean, dz = v.z - mean, dw = v.w - mean;
  red[tid] = dx * dx + dy * dy + dz * dz + dw * dw;
  __syncthreads();
  for (int st = 128; st > 0; st >>= 1) {
    if (tid < st) red[tid] += red[tid + st];
    __syncthreads();
  }
  float inv = rsqrtf(red[0] * (1.0f / Dc) + 1e-12f);

  int d = tid * 4;
  float o0 = gamma[d + 0] * dx * inv + beta[d + 0];
  float o1 = gamma[d + 1] * dy * inv + beta[d + 1];
  float o2 = gamma[d + 2] * dz * inv + beta[d + 2];
  float o3 = gamma[d + 3] * dw * inv + beta[d + 3];
  size_t base = (size_t)row * Dc + d;
  outf[base + 0] = o0; outf[base + 1] = o1;
  outf[base + 2] = o2; outf[base + 3] = o3;
  if constexpr (WRITE_BF16) {
    outb[base + 0] = (__bf16)o0; outb[base + 1] = (__bf16)o1;
    outb[base + 2] = (__bf16)o2; outb[base + 3] = (__bf16)o3;
  }
}

// ---------------------------------------------------------------------------
// Host orchestration
// ---------------------------------------------------------------------------
extern "C" void kernel_launch(void* const* d_in, const int* in_sizes, int n_in,
                              void* d_out, int out_size, void* d_ws, size_t ws_size,
                              hipStream_t stream) {
  const float* x         = (const float*)d_in[0];
  const float* attn_bias = (const float*)d_in[1];
  const int*   src_mask  = (const int*)d_in[2];
  const float* wq = (const float*)d_in[3];  const float* bq = (const float*)d_in[4];
  const float* wk = (const float*)d_in[5];  const float* bk = (const float*)d_in[6];
  const float* wv = (const float*)d_in[7];  const float* bv = (const float*)d_in[8];
  const float* wo = (const float*)d_in[9];  const float* bo = (const float*)d_in[10];
  const float* gamma1 = (const float*)d_in[11]; const float* beta1 = (const float*)d_in[12];
  const float* w1 = (const float*)d_in[13]; const float* b1 = (const float*)d_in[14];
  const float* w2 = (const float*)d_in[15]; const float* b2 = (const float*)d_in[16];
  const float* gamma2 = (const float*)d_in[17]; const float* beta2 = (const float*)d_in[18];

  char* ws = (char*)d_ws;
  size_t off = 0;
  auto alloc = [&](size_t bytes) -> void* {
    void* p = ws + off;
    off += (bytes + 255) & ~(size_t)255;
    return p;
  };
  constexpr size_t NTOK = (size_t)Mrows * Dc;      // 4M
  __bf16* xb   = (__bf16*)alloc(NTOK * 2);
  __bf16* wqb  = (__bf16*)alloc((size_t)Dc * Dc * 2);
  __bf16* wkb  = (__bf16*)alloc((size_t)Dc * Dc * 2);
  __bf16* wvb  = (__bf16*)alloc((size_t)Dc * Dc * 2);
  __bf16* wob  = (__bf16*)alloc((size_t)Dc * Dc * 2);
  __bf16* w1b  = (__bf16*)alloc((size_t)Dc * FFc * 2);
  __bf16* w2b  = (__bf16*)alloc((size_t)FFc * Dc * 2);
  __bf16* qb   = (__bf16*)alloc(NTOK * 2);
  __bf16* kb   = (__bf16*)alloc(NTOK * 2);
  __bf16* vb   = (__bf16*)alloc(NTOK * 2);
  __bf16* ctxb = (__bf16*)alloc(NTOK * 2);
  float*  y1   = (float*)alloc(NTOK * 4);          // reused as y2
  float*  hf   = (float*)alloc(NTOK * 4);
  __bf16* hb   = (__bf16*)alloc(NTOK * 2);
  __bf16* ub   = (__bf16*)alloc((size_t)Mrows * FFc * 2);
  (void)ws_size; (void)n_in; (void)in_sizes; (void)out_size;

  auto conv = [&](const float* src, __bf16* dst, int n) {
    f32_to_bf16_kernel<<<(n + 255) / 256, 256, 0, stream>>>(src, dst, n);
  };
  conv(x,  xb,  (int)NTOK);
  conv(wq, wqb, Dc * Dc);
  conv(wk, wkb, Dc * Dc);
  conv(wv, wvb, Dc * Dc);
  conv(wo, wob, Dc * Dc);
  conv(w1, w1b, Dc * FFc);
  conv(w2, w2b, FFc * Dc);

  // Q/K/V projections: [4096,1024] x [1024,1024] -> head-split bf16
  dim3 gQ(Dc / 128, Mrows / 64);
  gemm_bf16_kernel<EPI_QKV><<<gQ, 256, 0, stream>>>(xb, wqb, bq, nullptr, qb, Mrows, Dc, Dc);
  gemm_bf16_kernel<EPI_QKV><<<gQ, 256, 0, stream>>>(xb, wkb, bk, nullptr, kb, Mrows, Dc, Dc);
  gemm_bf16_kernel<EPI_QKV><<<gQ, 256, 0, stream>>>(xb, wvb, bv, nullptr, vb, Mrows, Dc, Dc);

  // flash attention -> merged-head context bf16 [B,L,D]
  flash_attn_kernel<<<dim3(Lc / 128, Hc, Bc), 256, 0, stream>>>(qb, kb, vb, attn_bias,
                                                                src_mask, ctxb);

  // output projection + residual(x): y1 = ctx @ wo + bo + x
  gemm_bf16_kernel<EPI_F32RES><<<gQ, 256, 0, stream>>>(ctxb, wob, bo, x, y1, Mrows, Dc, Dc);

  // LN1 -> h (f32 for residual) + hb (bf16 for FFN GEMM)
  layernorm_kernel<true><<<Mrows, 256, 0, stream>>>(y1, gamma1, beta1, hf, hb);

  // FFN1: u = relu(h @ w1 + b1), bf16  [4096,1024]x[1024,4096]
  dim3 gF1(FFc / 128, Mrows / 64);
  gemm_bf16_kernel<EPI_RELU_BF16><<<gF1, 256, 0, stream>>>(hb, w1b, b1, nullptr, ub,
                                                           Mrows, FFc, Dc);

  // FFN2: y2 = u @ w2 + b2 + h   [4096,4096]x[4096,1024]
  gemm_bf16_kernel<EPI_F32RES><<<gQ, 256, 0, stream>>>(ub, w2b, b2, hf, y1, Mrows, Dc, FFc);

  // LN2 -> final fp32 output
  layernorm_kernel<false><<<Mrows, 256, 0, stream>>>(y1, gamma2, beta2, (float*)d_out, nullptr);
}
